// TrajectoryPrediction_42649025249918
// MI455X (gfx1250) — matrix operand order repaired and checked by reading, not compiled, not run on hardware
//
#include <hip/hip_runtime.h>

typedef __attribute__((ext_vector_type(16))) _Float16 v16h;
typedef __attribute__((ext_vector_type(8)))  _Float16 v8h;
typedef __attribute__((ext_vector_type(8)))  float    v8f;
typedef __attribute__((ext_vector_type(4)))  float    v4f;

#define B_DIM 16384
#define D_DIM 64
#define H_DIM 64

// Single-instruction ReLU: v_med3_f32(x, 0, +inf) clamps to [0, inf).
__device__ __forceinline__ float relu1(float x) {
    return __builtin_amdgcn_fmed3f(x, 0.0f, __builtin_huge_valf());
}

// ---------------- prepass 1: X (f32) -> Xh (f16), row-major [B, D] ----------------
__global__ __launch_bounds__(256) void cvt_x_kernel(const float* __restrict__ X,
                                                    _Float16* __restrict__ Xh) {
    const int t = blockIdx.x * 256 + threadIdx.x;     // each thread: 8 elements
    v4f a = ((const v4f*)X)[2 * t];
    v4f b = ((const v4f*)X)[2 * t + 1];
    v8h o;
    #pragma unroll
    for (int e = 0; e < 4; ++e) {
        o[e]     = (_Float16)a[e];
        o[4 + e] = (_Float16)b[e];
    }
    ((v8h*)Xh)[t] = o;
}

// -------- prepass 2: W1eff[i,h,d] = (f16)(W1[i,h,d] * Adj[i,d]), row-major --------
__global__ __launch_bounds__(256) void cvt_w1_kernel(const float* __restrict__ W1,
                                                     const float* __restrict__ Adj,
                                                     _Float16* __restrict__ W1h) {
    const int t  = blockIdx.x * 256 + threadIdx.x;    // each thread: 8 elements
    const int i  = t >> 9;                            // node = (t*8) / (H*D)
    const int d0 = (t & 7) * 8;                       // d offset within row
    v4f w0 = ((const v4f*)W1)[2 * t];
    v4f w1 = ((const v4f*)W1)[2 * t + 1];
    v4f a0 = *(const v4f*)(Adj + i * D_DIM + d0);
    v4f a1 = *(const v4f*)(Adj + i * D_DIM + d0 + 4);
    v8h o;
    #pragma unroll
    for (int e = 0; e < 4; ++e) {
        o[e]     = (_Float16)(w0[e] * a0[e]);
        o[4 + e] = (_Float16)(w1[e] * a1[e]);
    }
    ((v8h*)W1h)[t] = o;
}

// ---------------- main kernel: pure-f16 fragment loads + WMMA ----------------
// One wave = one node i, 64 batch rows (4 M-tiles of 16).
__global__ __launch_bounds__(256) void traj_main_kernel(
    const _Float16* __restrict__ Xh,   // [B, D] f16
    const _Float16* __restrict__ W1h,  // [D, H, D] f16 (adjacency folded in)
    const float*    __restrict__ W2,   // [D, H] f32
    float*          __restrict__ out)  // [B, D] f32
{
    __shared__ float red[8][16 * 16];  // per-wave reduction tile

    const int tid   = threadIdx.x;
    const int wave  = tid >> 5;
    const int lane  = tid & 31;
    const int wid   = blockIdx.x * 8 + wave;       // 0 .. 16383
    const int node  = wid & (D_DIM - 1);
    const int mbase = (wid >> 6) * 64;
    const int hf    = lane >> 4;
    const int lmod  = lane & 15;

    // B fragments of W1eff[node]^T: nt = N-tile (h = nt*16 + lmod), kc = K-chunk.
    // B(32x16,f16), this code's k order (same as A): v16h element e <-> K = kc*32 + hf*16 + e (contiguous in d).
    v16h bfrag[4][2];
    {
        const _Float16* w1n = W1h + (size_t)node * H_DIM * D_DIM;
        #pragma unroll
        for (int nt = 0; nt < 4; ++nt) {
            const _Float16* wrow = w1n + (size_t)(nt * 16 + lmod) * D_DIM;
            #pragma unroll
            for (int kc = 0; kc < 2; ++kc) {
                const v8h* p = (const v8h*)(wrow + kc * 32 + hf * 16);
                v8h b0 = p[0], b1 = p[1];
                bfrag[nt][kc] = __builtin_shufflevector(
                    b0, b1, 0, 1, 2, 3, 4, 5, 6, 7, 8, 9, 10, 11, 12, 13, 14, 15);
            }
        }
    }

    float w2s[4];
    #pragma unroll
    for (int nt = 0; nt < 4; ++nt)
        w2s[nt] = W2[node * H_DIM + nt * 16 + lmod];

    #pragma unroll
    for (int mt = 0; mt < 4; ++mt) {
        const _Float16* xrow = Xh + (size_t)(mbase + mt * 16 + lmod) * D_DIM;

        // A(16x32,f16), this code's k order: elements 0..7  <-> K = kc*32 + hf*16 + [0,8)
        //                   elements 8..15 <-> K = kc*32 + hf*16 + 8 + [0,8)
        v16h afrag[2];
        #pragma unroll
        for (int kc = 0; kc < 2; ++kc) {
            v8h a0 = *(const v8h*)(xrow + kc * 32 + hf * 16);
            v8h a1 = *(const v8h*)(xrow + kc * 32 + hf * 16 + 8);
            afrag[kc] = __builtin_shufflevector(
                a0, a1, 0, 1, 2, 3, 4, 5, 6, 7, 8, 9, 10, 11, 12, 13, 14, 15);
        }

        float acc[8];
        #pragma unroll
        for (int r = 0; r < 8; ++r) acc[r] = 0.0f;

        #pragma unroll
        for (int nt = 0; nt < 4; ++nt) {
            v8f c = {};
            c = __builtin_amdgcn_wmma_f32_16x16x32_f16(
                    false, afrag[0], false, bfrag[nt][0], (short)0, c, false, false);
            c = __builtin_amdgcn_wmma_f32_16x16x32_f16(
                    false, afrag[1], false, bfrag[nt][1], (short)0, c, false, false);
            // C layout: lane -> N (= hidden col lmod), VGPR r -> M (= r + 8*hf)
            #pragma unroll
            for (int r = 0; r < 8; ++r)
                acc[r] += relu1(c[r]) * w2s[nt];          // ReLU * W2[i,h]
        }

        // intra-wave transpose + reduce over hidden index via wave-private LDS tile
        float* myred = &red[wave][0];
        #pragma unroll
        for (int r = 0; r < 8; ++r)
            myred[(r + 8 * hf) * 16 + lmod] = acc[r];     // [m][n], m=r+8hf, n=lmod
        // LDS ops of one wave are in-order: reads below see all lanes' writes.
        {
            v4f s0 = *(const v4f*)(myred + lmod * 16 + hf * 8);
            v4f s1 = *(const v4f*)(myred + lmod * 16 + hf * 8 + 4);
            float v = ((s0[0] + s0[1]) + (s0[2] + s0[3])) +
                      ((s1[0] + s1[1]) + (s1[2] + s1[3]));
            v += __shfl_xor(v, 16, 32);                   // combine the two halves
            if (hf == 0)
                out[(size_t)(mbase + mt * 16 + lmod) * D_DIM + node] = v;
        }
    }
}

// ---------------- fallback (no workspace): fused conversion version ----------------
__global__ __launch_bounds__(256) void traj_fallback_kernel(
    const float* __restrict__ X, const float* __restrict__ Adj,
    const float* __restrict__ W1, const float* __restrict__ W2,
    float* __restrict__ out)
{
    const int tid   = threadIdx.x;
    const int wave  = tid >> 5;
    const int lane  = tid & 31;
    const int wid   = blockIdx.x * 8 + wave;
    const int node  = wid & (D_DIM - 1);
    const int mbase = (wid >> 6) * 64;
    const int hf    = lane >> 4;
    const int lmod  = lane & 15;

    v4f adj[2][4];
    {
        const float* arow = Adj + node * D_DIM;
        #pragma unroll
        for (int kc = 0; kc < 2; ++kc)
            #pragma unroll
            for (int q = 0; q < 4; ++q)
                adj[kc][q] = *(const v4f*)(arow + kc * 32 + hf * 16 + 4 * q);
    }

    v16h bfrag[4][2];
    {
        const float* w1n = W1 + (size_t)node * H_DIM * D_DIM;
        #pragma unroll
        for (int nt = 0; nt < 4; ++nt) {
            const float* wrow = w1n + (size_t)(nt * 16 + lmod) * D_DIM;
            #pragma unroll
            for (int kc = 0; kc < 2; ++kc) {
                v16h bf;
                #pragma unroll
                for (int q = 0; q < 4; ++q) {
                    v4f w = *(const v4f*)(wrow + kc * 32 + hf * 16 + 4 * q);
                    #pragma unroll
                    for (int e = 0; e < 4; ++e)
                        bf[4 * q + e] = (_Float16)(w[e] * adj[kc][q][e]);
                }
                bfrag[nt][kc] = bf;
            }
        }
    }

    float w2s[4];
    #pragma unroll
    for (int nt = 0; nt < 4; ++nt)
        w2s[nt] = W2[node * H_DIM + nt * 16 + lmod];

    #pragma unroll
    for (int mt = 0; mt < 4; ++mt) {
        const float* xrow = X + (size_t)(mbase + mt * 16 + lmod) * D_DIM;
        v16h afrag[2];
        #pragma unroll
        for (int kc = 0; kc < 2; ++kc) {
            v16h af;
            #pragma unroll
            for (int q = 0; q < 2; ++q) {
                v4f x0 = *(const v4f*)(xrow + kc * 32 + hf * 16 + 4 * q);
                v4f x1 = *(const v4f*)(xrow + kc * 32 + hf * 16 + 8 + 4 * q);
                #pragma unroll
                for (int e = 0; e < 4; ++e) {
                    af[4 * q + e]     = (_Float16)x0[e];
                    af[8 + 4 * q + e] = (_Float16)x1[e];
                }
            }
            afrag[kc] = af;
        }

        float acc[8];
        #pragma unroll
        for (int r = 0; r < 8; ++r) acc[r] = 0.0f;

        #pragma unroll
        for (int nt = 0; nt < 4; ++nt) {
            v8f c = {};
            c = __builtin_amdgcn_wmma_f32_16x16x32_f16(
                    false, afrag[0], false, bfrag[nt][0], (short)0, c, false, false);
            c = __builtin_amdgcn_wmma_f32_16x16x32_f16(
                    false, afrag[1], false, bfrag[nt][1], (short)0, c, false, false);
            #pragma unroll
            for (int r = 0; r < 8; ++r)
                acc[r] += relu1(c[r]) * w2s[nt];
        }

        #pragma unroll
        for (int r = 0; r < 8; ++r) {
            float v = acc[r];
            v += __shfl_xor(v, 1, 32);
            v += __shfl_xor(v, 2, 32);
            v += __shfl_xor(v, 4, 32);
            v += __shfl_xor(v, 8, 32);
            acc[r] = v;
        }
        if (lmod == 0) {
            float* o = out + (size_t)(mbase + mt * 16 + 8 * hf) * D_DIM + node;
            #pragma unroll
            for (int r = 0; r < 8; ++r)
                o[(size_t)r * D_DIM] = acc[r];
        }
    }
}

extern "C" void kernel_launch(void* const* d_in, const int* in_sizes, int n_in,
                              void* d_out, int out_size, void* d_ws, size_t ws_size,
                              hipStream_t stream) {
    const float* X   = (const float*)d_in[0];  // inputs   [B, D]
    const float* Adj = (const float*)d_in[1];  // adjacency[D, D]
    const float* W1  = (const float*)d_in[2];  // W1       [D, H, D]
    const float* W2  = (const float*)d_in[3];  // W2       [D, H]
    float* out = (float*)d_out;                // [B, D]

    const size_t xh_bytes  = (size_t)B_DIM * D_DIM * sizeof(_Float16);          // 2 MB
    const size_t w1h_bytes = (size_t)D_DIM * H_DIM * D_DIM * sizeof(_Float16);  // 512 KB

    if (ws_size >= xh_bytes + w1h_bytes) {
        _Float16* Xh  = (_Float16*)d_ws;
        _Float16* W1h = (_Float16*)((char*)d_ws + xh_bytes);
        // 1M elems / 8 per thread = 131072 threads
        hipLaunchKernelGGL(cvt_x_kernel, dim3(512), dim3(256), 0, stream, X, Xh);
        // 256K elems / 8 per thread = 32768 threads
        hipLaunchKernelGGL(cvt_w1_kernel, dim3(128), dim3(256), 0, stream, W1, Adj, W1h);
        // waves = (B/64)*D = 16384 ; 8 waves / block
        hipLaunchKernelGGL(traj_main_kernel, dim3(2048), dim3(256), 0, stream,
                           Xh, W1h, W2, out);
    } else {
        hipLaunchKernelGGL(traj_fallback_kernel, dim3(2048), dim3(256), 0, stream,
                           X, Adj, W1, W2, out);
    }
}
